// Net_66159676227963
// MI455X (gfx1250) — matrix operand order repaired and checked by reading, not compiled, not run on hardware
//
#include <hip/hip_runtime.h>
#include <hip/hip_bf16.h>
#include <math.h>

// ---------------------------------------------------------------------------
// Problem constants (from reference)
// ---------------------------------------------------------------------------
#define NROWS 50000
#define IN_C  512
#define DIM   2048
#define OUT_C 256

typedef __attribute__((ext_vector_type(16))) __bf16       v16bf;
typedef __attribute__((ext_vector_type(8)))  float        v8f;
typedef __attribute__((ext_vector_type(4)))  unsigned int u32x4;

union ABFrag { u32x4 u[2]; v16bf v; };

// ---------------------------------------------------------------------------
// Prep kernels
// ---------------------------------------------------------------------------

// f32 -> bf16 row-major conversion of activations/input
__global__ void cvt_f32_bf16(const float* __restrict__ in,
                             unsigned short* __restrict__ out, size_t n) {
    for (size_t i = blockIdx.x * (size_t)blockDim.x + threadIdx.x; i < n;
         i += (size_t)gridDim.x * blockDim.x) {
        __bf16 h = (__bf16)in[i];
        out[i] = __builtin_bit_cast(unsigned short, h);
    }
}

// Fuse Wg+Wl (f32, row-major K x N) and pack into WMMA B-fragment layout:
// flat = ((kc*nColTiles + ct)*32 + lane)*16 + t
//   k = kc*32 + (lane>>4)*16 + t,  n = ct*16 + (lane&15)
// Contiguous k order (not AMD's table order), same as the A loads, for WMMA B (32x16, 16-bit): lanes 0-15 hold K 0..15,
// lanes 16-31 hold K 16..31, column = lane%16, K ascending within the lane's
// 32-byte chunk.
__global__ void pack_w(const float* __restrict__ Wg, const float* __restrict__ Wl,
                       unsigned short* __restrict__ out, int Kdim, int Ndim) {
    const size_t total = (size_t)Kdim * Ndim;
    const int nColTiles = Ndim >> 4;
    for (size_t e = blockIdx.x * (size_t)blockDim.x + threadIdx.x; e < total;
         e += (size_t)gridDim.x * blockDim.x) {
        const int t    = (int)(e & 15);
        const int lane = (int)((e >> 4) & 31);
        const size_t blk = e >> 9;                 // 512 elements per fragment
        const int ct = (int)(blk % nColTiles);
        const int kc = (int)(blk / nColTiles);
        const int k = kc * 32 + (lane >> 4) * 16 + t;
        const int n = ct * 16 + (lane & 15);
        const float v = Wg[(size_t)k * Ndim + n] + Wl[(size_t)k * Ndim + n];
        __bf16 h = (__bf16)v;
        out[e] = __builtin_bit_cast(unsigned short, h);
    }
}

// Fused bias: bg + bl (f32)
__global__ void fuse_bias(const float* __restrict__ bg, const float* __restrict__ bl,
                          float* __restrict__ out, int n) {
    int i = blockIdx.x * blockDim.x + threadIdx.x;
    if (i < n) out[i] = bg[i] + bl[i];
}

// ---------------------------------------------------------------------------
// WMMA GEMM:  out = act( A(bf16, nrows x Kdim) @ Wpacked(Kdim x Ndim) + bias )
// 256 threads = 8 wave32, workgroup tile 128 rows x 256 cols.
// Each wave: 4 row-tiles x 4 col-tiles of 16x16 f32 accumulators.
// ---------------------------------------------------------------------------
template <bool FINAL>
__global__ __launch_bounds__(256, 1)
void gemm_wmma_bf16(const unsigned short* __restrict__ A,
                    const unsigned short* __restrict__ Wp,
                    const float* __restrict__ bias,
                    void* __restrict__ outP,
                    int nrows, int Kdim, int Ndim) {
    const int lane    = threadIdx.x & 31;
    const int wave    = threadIdx.x >> 5;
    const int waveRow = wave >> 2;   // 0..1
    const int waveCol = wave & 3;    // 0..3
    const int laneRow = lane & 15;
    const int halfSel = lane >> 4;   // 0 or 1

    const int rowBase = blockIdx.x * 128 + waveRow * 64;
    const int colBase = blockIdx.y * 256 + waveCol * 64;
    const int nColTiles = Ndim >> 4;
    const int colTile0  = colBase >> 4;
    const int nK = Kdim >> 5;

    v8f acc[4][4] = {};

    // Per-row-tile A base (element units). Tiles are wave-uniformly valid
    // (nrows % 16 == 0), clamp invalid tiles to row 0 so EXEC stays all-1s
    // through the WMMAs (ISA requirement).
    size_t aRow[4];
#pragma unroll
    for (int rt = 0; rt < 4; ++rt) {
        int r = rowBase + rt * 16 + laneRow;
        if (r >= nrows) r = laneRow;             // clamped, stores skipped later
        aRow[rt] = (size_t)r * Kdim + halfSel * 16;
    }

    for (int kc = 0; kc < nK; ++kc) {
        // B fragments: contiguous 32B per lane thanks to pre-swizzled layout
        ABFrag b[4];
        const unsigned short* bk =
            Wp + ((size_t)kc * nColTiles + colTile0) * 512 + lane * 16;
#pragma unroll
        for (int ct = 0; ct < 4; ++ct) {
            const u32x4* p = (const u32x4*)(bk + (size_t)ct * 512);
            b[ct].u[0] = p[0];
            b[ct].u[1] = p[1];
        }

        const int kOff = kc * 32;
#pragma unroll
        for (int rt = 0; rt < 4; ++rt) {
            // A fragment: lanes 0-15 hold K 0..15, lanes 16-31 K 16..31
            ABFrag a;
            const unsigned short* ap = A + aRow[rt] + kOff;
            a.u[0] = *(const u32x4*)ap;
            a.u[1] = *(const u32x4*)(ap + 8);
            __builtin_prefetch(ap + 32, 0, 1);   // global_prefetch_b8: next K chunk
#pragma unroll
            for (int ct = 0; ct < 4; ++ct) {
                acc[rt][ct] = __builtin_amdgcn_wmma_f32_16x16x32_bf16(
                    false, a.v, false, b[ct].v, (short)0, acc[rt][ct],
                    false, false);
            }
        }
    }

    // Epilogue: bias (+ELU) and store.
    // C/D 16x16 f32 layout: col = colTileBase + lane%16,
    // row = rowTileBase + (lane<16 ? r : 8+r) for VGPR r.
#pragma unroll
    for (int ct = 0; ct < 4; ++ct) {
        const int col = colBase + ct * 16 + laneRow;
        const float bv = bias[col];
#pragma unroll
        for (int rt = 0; rt < 4; ++rt) {
            const int rowTile = rowBase + rt * 16;
            if (rowTile >= nrows) continue;      // wave-uniform (after WMMAs)
#pragma unroll
            for (int r = 0; r < 8; ++r) {
                const int row = rowTile + halfSel * 8 + r;
                float v = acc[rt][ct][r] + bv;
                if (FINAL) {
                    ((float*)outP)[(size_t)row * Ndim + col] = v;
                } else {
                    v = v > 0.0f ? v : (__expf(v) - 1.0f);   // ELU, alpha=1
                    __bf16 h = (__bf16)v;
                    ((unsigned short*)outP)[(size_t)row * Ndim + col] =
                        __builtin_bit_cast(unsigned short, h);
                }
            }
        }
    }
}

// ---------------------------------------------------------------------------
// Launch
// ---------------------------------------------------------------------------
extern "C" void kernel_launch(void* const* d_in, const int* in_sizes, int n_in,
                              void* d_out, int out_size, void* d_ws, size_t ws_size,
                              hipStream_t stream) {
    (void)in_sizes; (void)n_in; (void)out_size; (void)ws_size;

    const float* x = (const float*)d_in[0];
    // d_in[1] = edge_index (unused by K=1 ChebConv)
    const float* Wg[5], *bg[5], *Wl[5], *bl[5];
    for (int l = 0; l < 5; ++l) {
        Wg[l] = (const float*)d_in[2 + 4 * l + 0];
        bg[l] = (const float*)d_in[2 + 4 * l + 1];
        Wl[l] = (const float*)d_in[2 + 4 * l + 2];
        bl[l] = (const float*)d_in[2 + 4 * l + 3];
    }

    const int Kd[5] = {IN_C, DIM, DIM, DIM, DIM};
    const int Nd[5] = {DIM, DIM, DIM, DIM, OUT_C};

    // Workspace carve-up (all offsets 256B aligned)
    char* ws = (char*)d_ws;
    size_t off = 0;
    auto alloc = [&](size_t bytes) {
        char* p = ws + off;
        off += (bytes + 255) & ~(size_t)255;
        return p;
    };
    unsigned short* xbf = (unsigned short*)alloc((size_t)NROWS * IN_C * 2);
    unsigned short* hA  = (unsigned short*)alloc((size_t)NROWS * DIM * 2);
    unsigned short* hB  = (unsigned short*)alloc((size_t)NROWS * DIM * 2);
    unsigned short* Wp[5];
    float* bf[5];
    for (int l = 0; l < 5; ++l)
        Wp[l] = (unsigned short*)alloc((size_t)Kd[l] * Nd[l] * 2);
    for (int l = 0; l < 5; ++l)
        bf[l] = (float*)alloc((size_t)Nd[l] * 4);

    // --- prep ---
    {
        size_t n = (size_t)NROWS * IN_C;
        cvt_f32_bf16<<<2048, 256, 0, stream>>>(x, xbf, n);
    }
    for (int l = 0; l < 5; ++l) {
        pack_w<<<2048, 256, 0, stream>>>(Wg[l], Wl[l], Wp[l], Kd[l], Nd[l]);
        fuse_bias<<<(Nd[l] + 255) / 256, 256, 0, stream>>>(bg[l], bl[l], bf[l], Nd[l]);
    }

    // --- GEMM chain ---
    const int gx = (NROWS + 127) / 128;   // 391 row blocks (50000 % 16 == 0)
    dim3 blk(256);

    // L1: xbf (K=512) -> hA (ELU)
    gemm_wmma_bf16<false><<<dim3(gx, DIM / 256), blk, 0, stream>>>(
        xbf, Wp[0], bf[0], hA, NROWS, IN_C, DIM);
    // L2: hA -> hB
    gemm_wmma_bf16<false><<<dim3(gx, DIM / 256), blk, 0, stream>>>(
        hA, Wp[1], bf[1], hB, NROWS, DIM, DIM);
    // L3: hB -> hA
    gemm_wmma_bf16<false><<<dim3(gx, DIM / 256), blk, 0, stream>>>(
        hB, Wp[2], bf[2], hA, NROWS, DIM, DIM);
    // L4: hA -> hB
    gemm_wmma_bf16<false><<<dim3(gx, DIM / 256), blk, 0, stream>>>(
        hA, Wp[3], bf[3], hB, NROWS, DIM, DIM);
    // L5: hB -> d_out (f32, no activation)
    gemm_wmma_bf16<true><<<dim3(gx, OUT_C / 256), blk, 0, stream>>>(
        hB, Wp[4], bf[4], d_out, NROWS, DIM, OUT_C);
}
